// DGCNN6_homo4_19653770346933
// MI455X (gfx1250) — compile-verified, matrix operand order checked by reading, not run on hardware
//
#include <hip/hip_runtime.h>
#include <hip/hip_bf16.h>
#include <stdint.h>

// ---------------------------------------------------------------------------
// DGCNN (DynamicEdgeConv x3 + homophily) for MI455X / gfx1250, wave32 + WMMA.
// All matrix math on bf16 WMMA (f32 accumulate). Operand data is stored
// pre-swizzled in the CDNA5 WMMA VGPR layout so every lane loads its 16
// bf16 halves as one contiguous 32B vector load (b128 pairs) - no packing movs.
// ---------------------------------------------------------------------------

#define NB 64          // graphs
#define NN 1024        // nodes per graph
#define BN (NB * NN)   // total nodes
#define KG 95          // homophily knn k
#define KC 15          // edgeconv knn k

typedef __attribute__((ext_vector_type(16))) __bf16 v16bf;
typedef __attribute__((ext_vector_type(8)))  float  v8f;

__device__ inline unsigned short f2bf(float f) {
  union { float f; unsigned int u; } c; c.f = f;
  unsigned int u = c.u;
  unsigned int r = u + 0x7fffu + ((u >> 16) & 1u);   // round-to-nearest-even
  return (unsigned short)(r >> 16);
}
__device__ inline float bf2f(unsigned short h) {
  union { unsigned int u; float f; } c; c.u = ((unsigned int)h) << 16;
  return c.f;
}
__device__ inline float lrelu_f(float v) { return v >= 0.f ? v : 0.01f * v; }

__device__ inline v8f wmma_bf16(v16bf a, v16bf b, v8f c) {
  return __builtin_amdgcn_wmma_f32_16x16x32_bf16(false, a, false, b, (short)0, c,
                                                 false, false);
}

// ---- CDNA5 16-bit WMMA operand swizzle --------------------------------------
// A tile 16(M)x32(K): lane = m + 16*g with g=(k>>3)&1; half index i within the
// lane's 16 halves: i = (k&7) + 8*(k>>4).  B tile 32(K)x16(N): lane = n + 16*g,
// same i.  We store tiles lane-major: tile[lane*16 + i].
__device__ inline int swz_i(int k) { return (k & 7) + 8 * ((k >> 4) & 1); }
__device__ inline int swz_g(int k) { return (k >> 3) & 1; }

// element (node, c) in a tiled-A feature array; ld = padded width (32/64/224).
// 16-node groups, each group = (ld/32) k-tiles of 512 ushorts.
__device__ inline size_t ta_addr(size_t node, int c, int ld) {
  const size_t grp = node >> 4;
  const int m = (int)(node & 15);
  const int ks = c >> 5, k = c & 31;
  return grp * ((size_t)ld * 16) + (size_t)ks * 512 +
         (size_t)((m + 16 * swz_g(k)) * 16 + swz_i(k));
}
// element (k, n) in a tiled-B weight array with N output cols.
__device__ inline size_t tb_addr(int k, int n, int N) {
  const int kt = k >> 5, nt = n >> 4, kk = k & 31, nn = n & 15;
  return (size_t)(kt * (N / 16) + nt) * 512 +
         (size_t)((nn + 16 * swz_g(kk)) * 16 + swz_i(kk));
}

__device__ inline v16bf vload(const unsigned short* p) {
  return *(const v16bf*)p;
}

// ---------------------------------------------------------------------------
// Kernel: zero accumulators (graph-replay safe: must re-zero every call)
// ---------------------------------------------------------------------------
__global__ void zero_kern(float* dsum, float* homo, int nd, int nh) {
  int i = blockIdx.x * blockDim.x + threadIdx.x;
  if (i < nd) dsum[i] = 0.f;
  if (i < nh) homo[i] = 0.f;
}

// ---------------------------------------------------------------------------
// Kernel: pack xx=[tq,x,pos] -> f32 [BN,5] and tiled bf16 (width 32, zero pad)
// ---------------------------------------------------------------------------
__global__ void pack_kern(const float* x, const float* pos, const float* tq,
                          float* feat5, unsigned short* featbf) {
  int i = blockIdx.x * blockDim.x + threadIdx.x;
  if (i >= BN) return;
  float f[5];
  f[0] = tq[i]; f[1] = x[i];
  f[2] = pos[3 * i + 0]; f[3] = pos[3 * i + 1]; f[4] = pos[3 * i + 2];
#pragma unroll
  for (int c = 0; c < 5; ++c) feat5[(size_t)i * 5 + c] = f[c];
#pragma unroll
  for (int c = 0; c < 32; ++c)
    featbf[ta_addr((size_t)i, c, 32)] = (c < 5) ? f2bf(f[c]) : (unsigned short)0;
}

// ---------------------------------------------------------------------------
// Kernel: convert/pad row-major [K,N] f32 weight into tiled-B bf16 [Kpad,N]
// ---------------------------------------------------------------------------
__global__ void cvt_pad_kern(const float* src, unsigned short* dst,
                             int K, int N, int Kpad) {
  int i = blockIdx.x * blockDim.x + threadIdx.x;
  if (i >= Kpad * N) return;
  int k = i / N, n = i % N;
  dst[tb_addr(k, n, N)] = (k < K) ? f2bf(src[k * N + n]) : (unsigned short)0;
}

// ---------------------------------------------------------------------------
// Kernel: squared L2 norms of tiled bf16 rows
// ---------------------------------------------------------------------------
__global__ void norm_kern(const unsigned short* xb, int ld, int f, float* norms) {
  int i = blockIdx.x * blockDim.x + threadIdx.x;
  if (i >= BN) return;
  float s = 0.f;
  for (int c = 0; c < f; ++c) {
    float v = bf2f(xb[ta_addr((size_t)i, c, ld)]);
    s += v * v;
  }
  norms[i] = s;
}

// ---------------------------------------------------------------------------
// Kernel: kNN per graph via WMMA Gram tiles + iterative wave32 argmin top-k.
// Block = 256 thr (8 waves): 16 query rows x 1024 candidates in 64KB LDS.
//   HOMO=true : k=95, self-excluded, accumulate exact-equality counts
//   HOMO=false: k=15, self included, write neighbor indices (padded to 16)
// ---------------------------------------------------------------------------
template <int FPAD, int KSEL, bool EXCL, bool HOMO>
__global__ __launch_bounds__(256) void knn_kern(const unsigned short* featbf,
                                                const float* norms,
                                                const float* feat5,
                                                int* idx_out, float* homo_cnt) {
  __shared__ float dist[16 * NN];          // 64 KB
  const int b = blockIdx.y, q0 = blockIdx.x * 16;
  const int t = threadIdx.x, wv = t >> 5, lane = t & 31;
  const size_t rowbase = (size_t)b * NN;

  // query A tiles: hoisted, contiguous 32B per lane (tiled layout)
  v16bf aq[FPAD / 32];
#pragma unroll
  for (int ks = 0; ks < FPAD / 32; ++ks)
    aq[ks] = vload(featbf + (rowbase + q0) * FPAD + (size_t)ks * 512 +
                   (size_t)lane * 16);

  const int g = lane >> 4, n = lane & 15;
  const float nq[1] = {0.f};  (void)nq;

  // wave wv covers candidate columns [wv*128, wv*128+128)
  for (int ct = 0; ct < 8; ++ct) {
    const int c0 = wv * 128 + ct * 16;
    v8f acc = {};
#pragma unroll
    for (int ks = 0; ks < FPAD / 32; ++ks) {
      v16bf bb = vload(featbf + (rowbase + c0) * FPAD + (size_t)ks * 512 +
                       (size_t)lane * 16);                 // X^T tile
      acc = wmma_bf16(aq[ks], bb, acc);
    }
    const int col = c0 + n;
    const float nc = norms[rowbase + col];
#pragma unroll
    for (int r = 0; r < 8; ++r) {
      const int row = r + 8 * g;
      float d = norms[rowbase + q0 + row] + nc - 2.f * acc[r];
      if (EXCL && (col == q0 + row)) d += 1e10f;
      dist[row * NN + col] = d;
    }
  }
  __syncthreads();

  // selection: wave handles rows {wv, wv+8}
  for (int rr = 0; rr < 2; ++rr) {
    const int row = wv + rr * 8;
    const int q = q0 + row;
    float cnt0 = 0.f, cnt1 = 0.f, cnt2 = 0.f, cnt3 = 0.f, cnt4 = 0.f;
    for (int ts = 0; ts < KSEL; ++ts) {
      float best = 3.0e38f; int bi = 0x7fffffff;
      for (int j = lane; j < NN; j += 32) {
        float v = dist[row * NN + j];
        if (v < best) { best = v; bi = j; }
      }
#pragma unroll
      for (int off = 16; off > 0; off >>= 1) {
        float ov = __shfl_xor(best, off, 32);
        int   oi = __shfl_xor(bi, off, 32);
        if (ov < best || (ov == best && oi < bi)) { best = ov; bi = oi; }
      }
      if (lane == 0) dist[row * NN + bi] = 3.0e38f;
      asm volatile("s_wait_dscnt 0x0" ::: "memory");   // CDNA5 split DS counter
      if (HOMO) {
        if (lane == 0) {
          const float* fq = feat5 + (rowbase + q) * 5;
          const float* fj = feat5 + (rowbase + bi) * 5;
          cnt0 += (fq[0] == fj[0]) ? 1.f : 0.f;
          cnt1 += (fq[1] == fj[1]) ? 1.f : 0.f;
          cnt2 += (fq[2] == fj[2]) ? 1.f : 0.f;
          cnt3 += (fq[3] == fj[3]) ? 1.f : 0.f;
          cnt4 += (fq[4] == fj[4]) ? 1.f : 0.f;
        }
      } else {
        if (lane == 0) idx_out[(rowbase + q) * 16 + ts] = bi;
      }
    }
    if (HOMO) {
      if (lane == 0) {
        atomicAdd(&homo_cnt[b * 5 + 0], cnt0);
        atomicAdd(&homo_cnt[b * 5 + 1], cnt1);
        atomicAdd(&homo_cnt[b * 5 + 2], cnt2);
        atomicAdd(&homo_cnt[b * 5 + 3], cnt3);
        atomicAdd(&homo_cnt[b * 5 + 4], cnt4);
      }
    } else {
      if (lane == 0) idx_out[(rowbase + q) * 16 + 15] = q;   // self pad row
    }
  }
}

// ---------------------------------------------------------------------------
// Kernel: EdgeConv MLP per node. Block = 128 thr (4 waves), one node each.
// 16 edge-message rows (15 real + 1 pad) -> WMMA MLP -> masked row-sum.
// LDS tiles staged directly in WMMA-A swizzled layout.
// ---------------------------------------------------------------------------
template <int KMSG, int H1, int H2, int FIN, int LDIN>
__global__ __launch_bounds__(128) void edgeconv_kern(
    const unsigned short* fin, const int* idx,
    const unsigned short* W1, const float* b1,
    const unsigned short* W2, const float* b2,
    unsigned short* out) {
  __shared__ alignas(32) unsigned short Am[16 * KMSG];
  __shared__ alignas(32) unsigned short H1s[16 * H1];
  __shared__ int nbr[16];
  const int ng = blockIdx.x;                  // global node id
  const size_t gb = (size_t)(ng & ~(NN - 1)); // graph base node
  const int t = threadIdx.x, wv = t >> 5, lane = t & 31;

  if (t < 16) nbr[t] = idx[(size_t)ng * 16 + t];
  __syncthreads();

  // stage message rows [x_i, x_j - x_i, pad 0] into swizzled A layout
  for (int i = t; i < 16 * KMSG; i += 128) {
    const int e = i / KMSG, c = i % KMSG;
    float v = 0.f;
    if (c < FIN) {
      v = bf2f(fin[ta_addr((size_t)ng, c, LDIN)]);
    } else if (c < 2 * FIN) {
      const int cc = c - FIN;
      const size_t jg = gb + (size_t)nbr[e];
      v = bf2f(fin[ta_addr(jg, cc, LDIN)]) -
          bf2f(fin[ta_addr((size_t)ng, cc, LDIN)]);
    }
    const int kk = c & 31;
    Am[(c >> 5) * 512 + (e + 16 * swz_g(kk)) * 16 + swz_i(kk)] = f2bf(v);
  }
  __syncthreads();

  __builtin_prefetch(W2, 0, 2);

  // GEMM1: [16 x KMSG] x [KMSG x H1], bias + lrelu -> LDS (swizzled A layout)
  for (int nt = wv; nt < H1 / 16; nt += 4) {
    v8f c = {};
#pragma unroll
    for (int ks = 0; ks < KMSG / 32; ++ks) {
      v16bf a  = vload(Am + ks * 512 + lane * 16);
      v16bf bm = vload(W1 + (size_t)(ks * (H1 / 16) + nt) * 512 + lane * 16);
      c = wmma_bf16(a, bm, c);
    }
    const int col = nt * 16 + (lane & 15);
    const float bias = b1[col];
    const int g = lane >> 4;
    const int kk = col & 31, kst = col >> 5;
#pragma unroll
    for (int r = 0; r < 8; ++r) {
      const int row = r + 8 * g;
      H1s[kst * 512 + (row + 16 * swz_g(kk)) * 16 + swz_i(kk)] =
          f2bf(lrelu_f(c[r] + bias));
    }
  }
  __syncthreads();

  // GEMM2: [16 x H1] x [H1 x H2], bias + lrelu, sum rows 0..14 (exclude pad)
  for (int nt = wv; nt < H2 / 16; nt += 4) {
    v8f c = {};
#pragma unroll
    for (int ks = 0; ks < H1 / 32; ++ks) {
      v16bf a  = vload(H1s + ks * 512 + lane * 16);
      v16bf bm = vload(W2 + (size_t)(ks * (H2 / 16) + nt) * 512 + lane * 16);
      c = wmma_bf16(a, bm, c);
    }
    const int col = nt * 16 + (lane & 15);
    const float bias = b2[col];
    const int g = lane >> 4;
    float s = 0.f;
#pragma unroll
    for (int r = 0; r < 8; ++r) {
      float v = lrelu_f(c[r] + bias);
      if (!(g == 1 && r == 7)) s += v;     // row 15 = pad edge
    }
    s += __shfl_xor(s, 16, 32);            // combine rows 0-7 with 8-15
    if (lane < 16) out[ta_addr((size_t)ng, col, 64)] = f2bf(s);
  }
}

// ---------------------------------------------------------------------------
// Kernel: build cat = [feat(5), x1, x2, x3, pad(27)] tiled bf16, width 224
// ---------------------------------------------------------------------------
__global__ void cat_kern(const float* feat5, const unsigned short* x1,
                         const unsigned short* x2, const unsigned short* x3,
                         unsigned short* cat) {
  size_t i = (size_t)blockIdx.x * blockDim.x + threadIdx.x;
  if (i >= (size_t)BN * 224) return;
  const size_t node = i / 224;
  const int c = (int)(i % 224);
  unsigned short v = 0;
  if (c < 5)        v = f2bf(feat5[node * 5 + c]);
  else if (c < 69)  v = x1[ta_addr(node, c - 5, 64)];
  else if (c < 133) v = x2[ta_addr(node, c - 69, 64)];
  else if (c < 197) v = x3[ta_addr(node, c - 133, 64)];
  cat[ta_addr(node, c, 224)] = v;
}

// ---------------------------------------------------------------------------
// Kernel: fused node MLP lrelu(cat@W1+b1)@W2 and per-graph sum (atomic).
// Block = 256 thr (8 waves), 16 node rows (never straddles a graph).
// ---------------------------------------------------------------------------
__global__ __launch_bounds__(256) void l1_kern(const unsigned short* cat,
                                               const unsigned short* W1,
                                               const float* b1,
                                               const unsigned short* W2,
                                               float* dsum) {
  __shared__ alignas(32) unsigned short A[16 * 224];    // 7 KB, tiled layout
  __shared__ alignas(32) unsigned short H1s[16 * 512];  // 16 KB, tiled layout
  const size_t r0 = (size_t)blockIdx.x * 16;
  const int bgraph = (int)(r0 / NN);
  const int t = threadIdx.x, wv = t >> 5, lane = t & 31;

  // group block is already in tiled layout in global memory: straight copy
  {
    const uint4* src = (const uint4*)(cat + r0 * 224);
    uint4* dst = (uint4*)A;
    for (int i = t; i < (16 * 224) / 8; i += 256) dst[i] = src[i];
  }
  __syncthreads();

  __builtin_prefetch(W1, 0, 2);

  // GEMM1: [16x224] x [224x512] + bias + lrelu -> LDS (swizzled)
  for (int nt = wv; nt < 32; nt += 8) {
    v8f c = {};
#pragma unroll
    for (int ks = 0; ks < 7; ++ks) {
      v16bf a  = vload(A + ks * 512 + lane * 16);
      v16bf bm = vload(W1 + (size_t)(ks * 32 + nt) * 512 + lane * 16);
      c = wmma_bf16(a, bm, c);
    }
    const int col = nt * 16 + (lane & 15);
    const float bias = b1[col];
    const int g = lane >> 4;
    const int kk = col & 31, kst = col >> 5;
#pragma unroll
    for (int r = 0; r < 8; ++r) {
      const int row = r + 8 * g;
      H1s[kst * 512 + (row + 16 * swz_g(kk)) * 16 + swz_i(kk)] =
          f2bf(lrelu_f(c[r] + bias));
    }
  }
  __syncthreads();

  // GEMM2: [16x512] x [512x256], no bias (added post-mean), sum all 16 rows
  for (int nt = wv; nt < 16; nt += 8) {
    v8f c = {};
#pragma unroll
    for (int ks = 0; ks < 16; ++ks) {
      v16bf a  = vload(H1s + ks * 512 + lane * 16);
      v16bf bm = vload(W2 + (size_t)(ks * 16 + nt) * 512 + lane * 16);
      c = wmma_bf16(a, bm, c);
    }
    float s = 0.f;
#pragma unroll
    for (int r = 0; r < 8; ++r) s += c[r];
    s += __shfl_xor(s, 16, 32);
    if (lane < 16) atomicAdd(&dsum[bgraph * 256 + nt * 16 + lane], s);
  }
}

// ---------------------------------------------------------------------------
// Kernel: head. g=[mean+b2(256), homo(5)] -> lrelu -> 261x256 -> lrelu -> 256x3
// ---------------------------------------------------------------------------
__global__ __launch_bounds__(256) void head_kern(const float* dsum,
                                                 const float* homo_cnt,
                                                 const float* l1_b2,
                                                 const float* mW1, const float* mb1,
                                                 const float* mW2, const float* mb2,
                                                 float* out) {
  __shared__ float gs[261];
  __shared__ float hs[256];
  const int t = threadIdx.x;
  for (int b = 0; b < NB; ++b) {
    gs[t] = lrelu_f(dsum[b * 256 + t] * (1.f / NN) + l1_b2[t]);
    if (t < 5)
      gs[256 + t] = lrelu_f(homo_cnt[b * 5 + t] * (1.f / ((float)NN * KG)));
    __syncthreads();
    float acc = mb1[t];
    for (int i = 0; i < 261; ++i) acc += gs[i] * mW1[i * 256 + t];
    hs[t] = lrelu_f(acc);
    __syncthreads();
    if (t < 3) {
      float o = mb2[t];
      for (int i = 0; i < 256; ++i) o += hs[i] * mW2[i * 3 + t];
      out[b * 3 + t] = o;
    }
    __syncthreads();
  }
}

// ---------------------------------------------------------------------------
extern "C" void kernel_launch(void* const* d_in, const int* in_sizes, int n_in,
                              void* d_out, int out_size, void* d_ws, size_t ws_size,
                              hipStream_t stream) {
  const float* x_in  = (const float*)d_in[0];
  const float* pos   = (const float*)d_in[1];
  const float* tq    = (const float*)d_in[2];
  // d_in[3] = batch (implied by layout)
  const float* c1W1 = (const float*)d_in[4];  const float* c1b1 = (const float*)d_in[5];
  const float* c1W2 = (const float*)d_in[6];  const float* c1b2 = (const float*)d_in[7];
  const float* c2W1 = (const float*)d_in[8];  const float* c2b1 = (const float*)d_in[9];
  const float* c2W2 = (const float*)d_in[10]; const float* c2b2 = (const float*)d_in[11];
  const float* l1W1 = (const float*)d_in[12]; const float* l1b1 = (const float*)d_in[13];
  const float* l1W2 = (const float*)d_in[14]; const float* l1b2 = (const float*)d_in[15];
  const float* mW1  = (const float*)d_in[16]; const float* mb1  = (const float*)d_in[17];
  const float* mW2  = (const float*)d_in[18]; const float* mb2  = (const float*)d_in[19];
  float* out = (float*)d_out;

  // ---- carve workspace (256B aligned)
  char* base = (char*)d_ws;
  size_t off = 0;
  auto carve = [&](size_t bytes) -> char* {
    char* p = base + off;
    off = (off + bytes + 255) & ~(size_t)255;
    return p;
  };
  float*          feat5  = (float*)         carve((size_t)BN * 5 * 4);
  unsigned short* featbf = (unsigned short*)carve((size_t)BN * 32 * 2);
  float*          norms  = (float*)         carve((size_t)BN * 4);
  unsigned short* x1bf   = (unsigned short*)carve((size_t)BN * 64 * 2);
  unsigned short* x2bf   = (unsigned short*)carve((size_t)BN * 64 * 2);
  unsigned short* x3bf   = (unsigned short*)carve((size_t)BN * 64 * 2);
  int*            idx    = (int*)           carve((size_t)BN * 16 * 4);
  unsigned short* w1c1   = (unsigned short*)carve((size_t)32 * 64 * 2);
  unsigned short* w2c1   = (unsigned short*)carve((size_t)64 * 64 * 2);
  unsigned short* w1c2   = (unsigned short*)carve((size_t)128 * 128 * 2);
  unsigned short* w2c2   = (unsigned short*)carve((size_t)128 * 64 * 2);
  unsigned short* w1l1   = (unsigned short*)carve((size_t)224 * 512 * 2);
  unsigned short* w2l1   = (unsigned short*)carve((size_t)512 * 256 * 2);
  unsigned short* catbf  = (unsigned short*)carve((size_t)BN * 224 * 2);
  float*          dsum   = (float*)         carve((size_t)NB * 256 * 4);
  float*          homo   = (float*)         carve((size_t)NB * 5 * 4);
  (void)ws_size; (void)n_in; (void)in_sizes; (void)out_size;

  // ---- 0. zero accumulators (every call: safe for graph replay)
  zero_kern<<<(NB * 256 + 255) / 256, 256, 0, stream>>>(dsum, homo, NB * 256, NB * 5);

  // ---- 1. pack features + bf16 tiled weights
  pack_kern<<<(BN + 255) / 256, 256, 0, stream>>>(x_in, pos, tq, feat5, featbf);
  cvt_pad_kern<<<(32 * 64 + 255) / 256, 256, 0, stream>>>(c1W1, w1c1, 10, 64, 32);
  cvt_pad_kern<<<(64 * 64 + 255) / 256, 256, 0, stream>>>(c1W2, w2c1, 64, 64, 64);
  cvt_pad_kern<<<(128 * 128 + 255) / 256, 256, 0, stream>>>(c2W1, w1c2, 128, 128, 128);
  cvt_pad_kern<<<(128 * 64 + 255) / 256, 256, 0, stream>>>(c2W2, w2c2, 128, 64, 128);
  cvt_pad_kern<<<(224 * 512 + 255) / 256, 256, 0, stream>>>(l1W1, w1l1, 197, 512, 224);
  cvt_pad_kern<<<(512 * 256 + 255) / 256, 256, 0, stream>>>(l1W2, w2l1, 512, 256, 512);

  const dim3 knn_grid(NN / 16, NB);

  // ---- 2. homophily kNN (k=95, exclude self) on 5-dim features
  norm_kern<<<(BN + 255) / 256, 256, 0, stream>>>(featbf, 32, 5, norms);
  knn_kern<32, KG, true, true><<<knn_grid, 256, 0, stream>>>(
      featbf, norms, feat5, idx, homo);

  // ---- 3. conv1: kNN(k=15) on feat, MLP 10->64->64, sum
  knn_kern<32, KC, false, false><<<knn_grid, 256, 0, stream>>>(
      featbf, norms, feat5, idx, homo);
  edgeconv_kern<32, 64, 64, 5, 32><<<BN, 128, 0, stream>>>(
      featbf, idx, w1c1, c1b1, w2c1, c1b2, x1bf);

  // ---- 4. conv2: kNN on x1, MLP 128->128->64
  norm_kern<<<(BN + 255) / 256, 256, 0, stream>>>(x1bf, 64, 64, norms);
  knn_kern<64, KC, false, false><<<knn_grid, 256, 0, stream>>>(
      x1bf, norms, feat5, idx, homo);
  edgeconv_kern<128, 128, 64, 64, 64><<<BN, 128, 0, stream>>>(
      x1bf, idx, w1c2, c2b1, w2c2, c2b2, x2bf);

  // ---- 5. conv3: same weights as conv2, on x2
  norm_kern<<<(BN + 255) / 256, 256, 0, stream>>>(x2bf, 64, 64, norms);
  knn_kern<64, KC, false, false><<<knn_grid, 256, 0, stream>>>(
      x2bf, norms, feat5, idx, homo);
  edgeconv_kern<128, 128, 64, 64, 64><<<BN, 128, 0, stream>>>(
      x2bf, idx, w1c2, c2b1, w2c2, c2b2, x3bf);

  // ---- 6. cat + fused node MLP (197->512->256) + per-graph sum
  cat_kern<<<(int)(((size_t)BN * 224 + 255) / 256), 256, 0, stream>>>(
      feat5, x1bf, x2bf, x3bf, catbf);
  l1_kern<<<BN / 16, 256, 0, stream>>>(catbf, w1l1, l1b1, w2l1, dsum);

  // ---- 7. head: [mean(256)+b2, homo(5)] -> 261->256->3
  head_kern<<<1, 256, 0, stream>>>(dsum, homo, l1b2, mW1, mb1, mW2, mb2, out);
}
